// DifferentiablePatchGrouping_45844480918007
// MI455X (gfx1250) — matrix operand order repaired and checked by reading, not compiled, not run on hardware
//
#include <hip/hip_runtime.h>
#include <stdint.h>
#include <stddef.h>

// ---------------------------------------------------------------------------
// MI455X / gfx1250 implementation.
// Compute-bound on conv2/conv3 (~194 GFLOPs total vs ~10MB external I/O), so
// all convs + linear layers are mapped to v_wmma_f32_16x16x32_bf16 implicit
// GEMM. Wave32 everywhere; conv input staged into LDS via the Tensor Data
// Mover (TENSOR_LOAD_TO_LDS + s_wait_tensorcnt) when available; weights are
// pre-packed to the WMMA A/B fragment lane orders the kernels load.
// ---------------------------------------------------------------------------

typedef __attribute__((ext_vector_type(16))) __bf16 v16bf;
typedef __attribute__((ext_vector_type(8)))  float  v8f;
typedef unsigned int u32x4 __attribute__((ext_vector_type(4)));
typedef int          i32x4 __attribute__((ext_vector_type(4)));
typedef int          i32x8 __attribute__((ext_vector_type(8)));

#if defined(__has_builtin)
#if __has_builtin(__builtin_amdgcn_tensor_load_to_lds) && \
    __has_builtin(__builtin_amdgcn_s_wait_tensorcnt)
#define USE_TDM 1
#endif
#endif
#ifndef USE_TDM
#define USE_TDM 0
#endif

union Frag {
  v16bf v;
  unsigned short u[16];
  uint4 q[2];
};

#define DEV __device__ __forceinline__

DEV unsigned short f2bf(float f) {
  unsigned int u = __float_as_uint(f);
  unsigned int r = (u + 0x7FFFu + ((u >> 16) & 1u)) >> 16;  // RNE
  return (unsigned short)r;
}
DEV float bf2f(unsigned short h) { return __uint_as_float(((unsigned int)h) << 16); }

static constexpr int IMG = 4096;  // B*N patches
static constexpr int PIX = 256;   // 16x16

// ---------------------------------------------------------------------------
// Weight pre-pack: conv weights -> WMMA A-fragment order.
// A 16x32 bf16 layout: lane L holds row M = L%16; element e -> K offset
//   ((L>>4)<<3) + (e<8 ? e : e+8)   within the 32-wide K slice.
// ---------------------------------------------------------------------------
template<int CIN, int COUT, int KST>
__global__ void prep_conv_w_kernel(const float* __restrict__ w,
                                   unsigned short* __restrict__ aprep) {
  const int i = blockIdx.x * 256 + threadIdx.x;
  constexpr int TOT = (COUT / 16) * KST * 512;
  if (i >= TOT) return;
  const int e = i & 15, lane = (i >> 4) & 31;
  const int kk = (i >> 9) % KST, omt = (i >> 9) / KST;
  const int oc = omt * 16 + (lane & 15);
  const int k = kk * 32 + ((lane >> 4) << 3) + ((e < 8) ? e : e + 8);
  float v = 0.f;
  if (k < CIN * 9) {
    const int ic = k / 9, tap = k - ic * 9;
    v = w[(oc * CIN + ic) * 9 + tap];
  }
  aprep[i] = f2bf(v);
}

// Linear weights (row-major [NOUT][K]) -> WMMA B-fragment order.
// B packing used here (contiguous k, same as gemm A loads): lane L holds col N = L%16; element e -> K offset
//   ((L>>4)<<4) + e  within the 32-wide K slice.
template<int K, int NT, int NREAL, int KST>
__global__ void prep_lin_w_kernel(const float* __restrict__ w,
                                  unsigned short* __restrict__ bp) {
  const int i = blockIdx.x * 256 + threadIdx.x;
  constexpr int TOT = NT * KST * 512;
  if (i >= TOT) return;
  const int e = i & 15, lane = (i >> 4) & 31;
  const int kk = (i >> 9) % KST, nt = (i >> 9) / KST;
  const int n = nt * 16 + (lane & 15);
  const int k = kk * 32 + ((lane >> 4) << 4) + e;
  const float v = (n < NREAL) ? w[n * K + k] : 0.f;
  bp[i] = f2bf(v);
}

// ---------------------------------------------------------------------------
// Implicit-GEMM conv: one block per patch; unpadded CIN x 16 x 16 image in
// LDS (staged via TDM when available); boundary-checked im2col gather; 8
// waves, wave w handles pixel tiles {w, w+8}, all COUT/16 M-tiles in regs.
// Raw (pre-BN) bf16 output; conv bias omitted (cancelled by train-mode BN).
// ---------------------------------------------------------------------------
template<int CIN, int COUT, int KST, bool INF32>
__global__ __launch_bounds__(256) void conv_wmma_kernel(
    const void* __restrict__ in_, const unsigned short* __restrict__ aprep,
    unsigned short* __restrict__ out) {
  constexpr int OMT = COUT / 16;
  constexpr int KREAL = CIN * 9;
  __shared__ unsigned short smem[CIN * PIX];
  const int img = blockIdx.x;
  const int tid = threadIdx.x;

  if (INF32) {
    const float* x = (const float*)in_;
    for (int i = tid; i < CIN * PIX; i += 256)
      smem[i] = f2bf(x[(size_t)img * CIN * PIX + i]);
  } else {
#if USE_TDM
    if (tid < 32) {  // one wave issues the tensor DMA (EXEC-independent op)
      const unsigned long long ga =
          (unsigned long long)(uintptr_t)in_ + (unsigned long long)img * (CIN * PIX * 2);
      const unsigned int lds = (unsigned int)(uintptr_t)&smem[0];  // addr[31:0]
      constexpr unsigned int ELEMS = (unsigned int)(CIN * PIX);    // bf16 elems
      // D# group0: count=1 | lds_addr | global_addr[56:0] | type=2
      u32x4 g0;
      g0[0] = 1u;
      g0[1] = lds;
      g0[2] = (unsigned int)ga;
      g0[3] = (unsigned int)((ga >> 32) & 0x1FFFFFFull) | (2u << 30);
      // D# group1: data_size=2B; tensor/tile = one row of ELEMS elements
      i32x8 g1;
      g1[0] = (int)(1u << 16);                    // data_size=1 (2 bytes)
      g1[1] = (int)((ELEMS & 0xFFFFu) << 16);     // tensor_dim0[15:0]
      g1[2] = (int)((ELEMS >> 16) | (1u << 16));  // tensor_dim0[31:16] | tensor_dim1=1
      g1[3] = (int)((ELEMS & 0xFFFFu) << 16);     // tile_dim0 = ELEMS
      g1[4] = 1;                                  // tile_dim1=1, tile_dim2=0
      g1[5] = (int)ELEMS;                         // tensor_dim0_stride[31:0]
      g1[6] = (int)((ELEMS & 0xFFFFu) << 16);     // stride hi=0 | dim1_stride[15:0]
      g1[7] = (int)(ELEMS >> 16);                 // dim1_stride[47:16]
      i32x4 z4 = {0, 0, 0, 0};
#if defined(__clang_major__) && (__clang_major__ >= 23)
      i32x8 z8 = {};
      __builtin_amdgcn_tensor_load_to_lds(g0, g1, z4, z4, z8, 0);
#else
      __builtin_amdgcn_tensor_load_to_lds(g0, g1, z4, z4, 0);
#endif
      __builtin_amdgcn_s_wait_tensorcnt(0);
    }
#else
    const unsigned short* a = (const unsigned short*)in_;
    for (int i = tid; i < CIN * PIX; i += 256)
      smem[i] = a[(size_t)img * CIN * PIX + i];
#endif
  }
  __syncthreads();

  const int wid = tid >> 5, lane = tid & 31;
  const int col = lane & 15;
  const int khalf = lane >> 4;
  const uint4* aq = (const uint4*)aprep;
  const v8f zero = {};

  for (int rep = 0; rep < 2; ++rep) {
    const int on = wid + (rep << 3);
    const int p = on * 16 + col;
    const int py = p >> 4, px = p & 15;
    v8f acc[OMT];
#pragma unroll
    for (int om = 0; om < OMT; ++om) acc[om] = zero;

    for (int kk = 0; kk < KST; ++kk) {
      Frag b;
      const int kb = kk * 32 + (khalf << 3);
#pragma unroll
      for (int e = 0; e < 16; ++e) {
        const int k = kb + ((e < 8) ? e : e + 8);
        unsigned short val = 0;
        if (k < KREAL) {
          const int ic = k / 9, tap = k - ic * 9;
          const int ty = tap / 3, tx = tap - ty * 3;
          const int iy = py + ty - 1, ix = px + tx - 1;
          if (((unsigned)iy < 16u) && ((unsigned)ix < 16u))
            val = smem[(ic << 8) + (iy << 4) + ix];
        }
        b.u[e] = val;
      }
#pragma unroll
      for (int om = 0; om < OMT; ++om) {
        Frag a;
        const uint4* src = aq + (((om * KST + kk) * 32 + lane) << 1);
        a.q[0] = src[0];
        a.q[1] = src[1];
        acc[om] = __builtin_amdgcn_wmma_f32_16x16x32_bf16(
            false, a.v, false, b.v, (short)0, acc[om], false, false);
      }
    }
    const int mrow = khalf << 3;
#pragma unroll
    for (int om = 0; om < OMT; ++om)
#pragma unroll
      for (int r = 0; r < 8; ++r) {
        const int oc = om * 16 + mrow + r;
        out[((size_t)img * COUT + oc) * PIX + p] = f2bf(acc[om][r]);
      }
  }
}

// ---------------------------------------------------------------------------
// Deterministic BN statistics: 64 partial chunks per channel, then finalize.
// ---------------------------------------------------------------------------
template<int C>
__global__ __launch_bounds__(256) void reduce_stats_kernel(
    const unsigned short* __restrict__ raw, float* __restrict__ psum,
    float* __restrict__ psq) {
  __shared__ float ss[256], sq[256];
  const int c = blockIdx.x >> 6;
  const int chunk = blockIdx.x & 63;
  const int tid = threadIdx.x;
  float s = 0.f, q = 0.f;
  const int base = chunk * 16384;
  for (int i = 0; i < 64; ++i) {
    const int t = base + tid + i * 256;  // < 1,048,576
    const int im = t >> 8, p = t & 255;
    const float v = bf2f(raw[((size_t)im * C + c) * 256 + p]);
    s += v;
    q += v * v;
  }
  ss[tid] = s;
  sq[tid] = q;
  __syncthreads();
  for (int off = 128; off > 0; off >>= 1) {
    if (tid < off) { ss[tid] += ss[tid + off]; sq[tid] += sq[tid + off]; }
    __syncthreads();
  }
  if (tid == 0) { psum[c * 64 + chunk] = ss[0]; psq[c * 64 + chunk] = sq[0]; }
}

template<int C>
__global__ void finalize_stats_kernel(const float* __restrict__ psum,
                                      const float* __restrict__ psq,
                                      const float* __restrict__ gamma,
                                      const float* __restrict__ beta,
                                      float* __restrict__ scale,
                                      float* __restrict__ shift) {
  const int c = threadIdx.x;
  if (c >= C) return;
  float s = 0.f, q = 0.f;
  for (int i = 0; i < 64; ++i) { s += psum[c * 64 + i]; q += psq[c * 64 + i]; }
  const float inv = 1.0f / 1048576.0f;
  const float m = s * inv;
  const float var = fmaxf(q * inv - m * m, 0.f);
  const float sc = gamma[c] * rsqrtf(var + 1e-5f);
  scale[c] = sc;
  shift[c] = beta[c] - m * sc;
}

template<int C>
__global__ __launch_bounds__(256) void bn_relu_kernel(
    unsigned short* __restrict__ data, const float* __restrict__ scale,
    const float* __restrict__ shift) {
  const int idx = blockIdx.x * 256 + threadIdx.x;  // < IMG*C*256
  const int c = (idx >> 8) % C;
  const float v = bf2f(data[idx]) * scale[c] + shift[c];
  data[idx] = f2bf(fmaxf(v, 0.f));
}

// Global average pool over 16x16 -> feat bf16 [IMG][128]
__global__ __launch_bounds__(128) void pool_kernel(
    const unsigned short* __restrict__ a3, unsigned short* __restrict__ feat) {
  const int img = blockIdx.x, c = threadIdx.x;
  const unsigned short* p = a3 + ((size_t)img * 128 + c) * 256;
  float s = 0.f;
  for (int i = 0; i < 256; ++i) s += bf2f(p[i]);
  feat[img * 128 + c] = f2bf(s * (1.0f / 256.0f));
}

// ---------------------------------------------------------------------------
// Generic WMMA GEMM: C[M=4096][N] = A[M][K](bf16,row-major) x Bprep + bias.
// One 16x16 output tile per wave, 8 waves per block.
// ---------------------------------------------------------------------------
template<int KST, int NT, int NREAL, bool OUTBF>
__global__ __launch_bounds__(256) void gemm_wmma_kernel(
    const unsigned short* __restrict__ A, const unsigned short* __restrict__ Bp,
    const float* __restrict__ bias, void* __restrict__ out_) {
  constexpr int K = KST * 32;
  constexpr int N = NT * 16;
  const int lane = threadIdx.x & 31;
  const int wid = threadIdx.x >> 5;
  const int t = blockIdx.x * 8 + wid;  // tile id over 256*NT tiles
  const int mt = t / NT, nt = t % NT;
  const int col = lane & 15;
  const int m = mt * 16 + col;
  const int k0 = (lane >> 4) << 4;
  const uint4* bq = (const uint4*)Bp;
  v8f acc = {};
  for (int kk = 0; kk < KST; ++kk) {
    Frag a, b;
    const unsigned short* ap = A + (size_t)m * K + kk * 32 + k0;
    a.q[0] = *(const uint4*)(ap);
    a.q[1] = *(const uint4*)(ap + 8);
    const uint4* bs = bq + (((nt * KST + kk) * 32 + lane) << 1);
    b.q[0] = bs[0];
    b.q[1] = bs[1];
    if (kk + 1 < KST) __builtin_prefetch(ap + 32, 0, 1);  // global_prefetch_b8
    acc = __builtin_amdgcn_wmma_f32_16x16x32_bf16(false, a.v, false, b.v,
                                                  (short)0, acc, false, false);
  }
  const int n = nt * 16 + col;
  const float bv = (n < NREAL) ? bias[n] : 0.f;
  const int mrow = (lane >> 4) << 3;
  if (OUTBF) {
    unsigned short* out = (unsigned short*)out_;
#pragma unroll
    for (int r = 0; r < 8; ++r)
      out[(size_t)(mt * 16 + mrow + r) * N + n] = f2bf(acc[r] + bv);
  } else {
    float* out = (float*)out_;
#pragma unroll
    for (int r = 0; r < 8; ++r)
      out[(size_t)(mt * 16 + mrow + r) * N + n] = acc[r] + bv;
  }
}

// LayerNorm (dim 512) + ReLU, f32 in -> bf16 out
__global__ __launch_bounds__(256) void ln_relu_kernel(
    const float* __restrict__ x, const float* __restrict__ g,
    const float* __restrict__ b, unsigned short* __restrict__ y) {
  __shared__ float ss[256], sq[256];
  const int row = blockIdx.x, tid = threadIdx.x;
  const float* xr = x + (size_t)row * 512;
  const float v0 = xr[tid], v1 = xr[tid + 256];
  ss[tid] = v0 + v1;
  sq[tid] = v0 * v0 + v1 * v1;
  __syncthreads();
  for (int off = 128; off > 0; off >>= 1) {
    if (tid < off) { ss[tid] += ss[tid + off]; sq[tid] += sq[tid + off]; }
    __syncthreads();
  }
  const float m = ss[0] * (1.0f / 512.0f);
  const float var = fmaxf(sq[0] * (1.0f / 512.0f) - m * m, 0.f);
  const float inv = rsqrtf(var + 1e-5f);
  unsigned short* yr = y + (size_t)row * 512;
  yr[tid] = f2bf(fmaxf((v0 - m) * inv * g[tid] + b[tid], 0.f));
  yr[tid + 256] = f2bf(fmaxf((v1 - m) * inv * g[tid + 256] + b[tid + 256], 0.f));
}

// logits/TEMP -> softmax over G=8; writes assignment (d_out) and w^T (ws)
__global__ __launch_bounds__(256) void softmax_kernel(
    const float* __restrict__ g3out, float* __restrict__ assign_out,
    float* __restrict__ wt) {
  const int row = blockIdx.x * 256 + threadIdx.x;  // < 4096
  const float* l = g3out + row * 16;
  float lv[8], mx = -1e30f;
#pragma unroll
  for (int j = 0; j < 8; ++j) { lv[j] = l[j] * 2.0f; mx = fmaxf(mx, lv[j]); }
  float s = 0.f;
#pragma unroll
  for (int j = 0; j < 8; ++j) { lv[j] = __expf(lv[j] - mx); s += lv[j]; }
  const float inv = 1.0f / s;
  const int bb = row >> 8, n = row & 255;
#pragma unroll
  for (int j = 0; j < 8; ++j) {
    const float a = lv[j] * inv;
    assign_out[row * 8 + j] = a;
    wt[((bb * 8 + j) << 8) + n] = a;
  }
}

// Stable rank-based top-32 of 256 (matches jax top_k tie-breaking)
__global__ __launch_bounds__(256) void topk_kernel(const float* __restrict__ wt,
                                                   float* __restrict__ topv,
                                                   int* __restrict__ topi) {
  __shared__ float w[256];
  const int bg = blockIdx.x, i = threadIdx.x;
  const float v = wt[bg * 256 + i];
  w[i] = v;
  __syncthreads();
  int rank = 0;
  for (int j = 0; j < 256; ++j) {
    const float wj = w[j];
    rank += (wj > v) || (wj == v && j < i);
  }
  if (rank < 32) {
    topv[bg * 32 + rank] = v;
    topi[bg * 32 + rank] = i;
  }
}

// output[b][g][k] = x[b][idx] * val
__global__ __launch_bounds__(256) void gather_kernel(
    const float* __restrict__ x, const float* __restrict__ topv,
    const int* __restrict__ topi, float* __restrict__ out) {
  const int bk = blockIdx.x;  // bg*32 + k, < 4096
  const int bg = bk >> 5;
  const int b = bg >> 3;
  const float val = topv[bk];
  const int idx = topi[bk];
  out[(size_t)bk * 256 + threadIdx.x] =
      x[(size_t)(b * 256 + idx) * 256 + threadIdx.x] * val;
}

// ---------------------------------------------------------------------------
extern "C" void kernel_launch(void* const* d_in, const int* in_sizes, int n_in,
                              void* d_out, int out_size, void* d_ws,
                              size_t ws_size, hipStream_t stream) {
  (void)in_sizes; (void)n_in; (void)out_size; (void)ws_size;
  const float* x       = (const float*)d_in[0];
  const float* conv1_w = (const float*)d_in[1];
  const float* bn1_g   = (const float*)d_in[3];
  const float* bn1_b   = (const float*)d_in[4];
  const float* conv2_w = (const float*)d_in[5];
  const float* bn2_g   = (const float*)d_in[7];
  const float* bn2_b   = (const float*)d_in[8];
  const float* conv3_w = (const float*)d_in[9];
  const float* bn3_g   = (const float*)d_in[11];
  const float* bn3_b   = (const float*)d_in[12];
  const float* fc_w    = (const float*)d_in[13];
  const float* fc_b    = (const float*)d_in[14];
  const float* g1_w    = (const float*)d_in[15];
  const float* g1_b    = (const float*)d_in[16];
  const float* ln1_g   = (const float*)d_in[17];
  const float* ln1_b   = (const float*)d_in[18];
  const float* g2_w    = (const float*)d_in[19];
  const float* g2_b    = (const float*)d_in[20];
  const float* ln2_g   = (const float*)d_in[21];
  const float* ln2_b   = (const float*)d_in[22];
  const float* g3_w    = (const float*)d_in[23];
  const float* g3_b    = (const float*)d_in[24];

  char* ws = (char*)d_ws;
  size_t off = 0;
  auto alloc = [&](size_t bytes) -> void* {
    void* p = ws + off;
    off += (bytes + 255) & ~(size_t)255;
    return p;
  };

  unsigned short* r1 = (unsigned short*)alloc((size_t)IMG * 32 * 256 * 2);
  unsigned short* r2 = (unsigned short*)alloc((size_t)IMG * 64 * 256 * 2);
  unsigned short* r3 = (unsigned short*)alloc((size_t)IMG * 128 * 256 * 2);
  unsigned short* ap1 = (unsigned short*)alloc(2 * 1 * 512 * 2);
  unsigned short* ap2 = (unsigned short*)alloc(4 * 9 * 512 * 2);
  unsigned short* ap3 = (unsigned short*)alloc(8 * 18 * 512 * 2);
  unsigned short* feat  = (unsigned short*)alloc((size_t)IMG * 128 * 2);
  unsigned short* bfc   = (unsigned short*)alloc(16 * 4 * 512 * 2);
  unsigned short* hfeat = (unsigned short*)alloc((size_t)IMG * 256 * 2);
  unsigned short* bg1   = (unsigned short*)alloc(32 * 8 * 512 * 2);
  float*          g1out = (float*)alloc((size_t)IMG * 512 * 4);
  unsigned short* h1    = (unsigned short*)alloc((size_t)IMG * 512 * 2);
  unsigned short* bg2   = (unsigned short*)alloc(32 * 16 * 512 * 2);
  float*          g2out = (float*)alloc((size_t)IMG * 512 * 4);
  unsigned short* h2    = (unsigned short*)alloc((size_t)IMG * 512 * 2);
  unsigned short* bg3   = (unsigned short*)alloc(1 * 16 * 512 * 2);
  float*          g3out = (float*)alloc((size_t)IMG * 16 * 4);
  float* wt   = (float*)alloc(16 * 8 * 256 * 4);
  float* topv = (float*)alloc(16 * 8 * 32 * 4);
  int*   topi = (int*)alloc(16 * 8 * 32 * 4);
  float* psum = (float*)alloc(128 * 64 * 4);
  float* psq  = (float*)alloc(128 * 64 * 4);
  float* bsc  = (float*)alloc(128 * 4);
  float* bsh  = (float*)alloc(128 * 4);

  float* out_main   = (float*)d_out;                         // (16,8,32,1,16,16)
  float* out_assign = out_main + (size_t)16 * 8 * 32 * 256;  // (16,256,8)

  // ---- weight pre-pack ----
  prep_conv_w_kernel<1, 32, 1><<<4, 256, 0, stream>>>(conv1_w, ap1);
  prep_conv_w_kernel<32, 64, 9><<<72, 256, 0, stream>>>(conv2_w, ap2);
  prep_conv_w_kernel<64, 128, 18><<<288, 256, 0, stream>>>(conv3_w, ap3);
  prep_lin_w_kernel<128, 16, 256, 4><<<128, 256, 0, stream>>>(fc_w, bfc);
  prep_lin_w_kernel<256, 32, 512, 8><<<512, 256, 0, stream>>>(g1_w, bg1);
  prep_lin_w_kernel<512, 32, 512, 16><<<1024, 256, 0, stream>>>(g2_w, bg2);
  prep_lin_w_kernel<512, 1, 8, 16><<<32, 256, 0, stream>>>(g3_w, bg3);

  // ---- conv1 + BN1 + ReLU ----
  conv_wmma_kernel<1, 32, 1, true><<<IMG, 256, 0, stream>>>(x, ap1, r1);
  reduce_stats_kernel<32><<<32 * 64, 256, 0, stream>>>(r1, psum, psq);
  finalize_stats_kernel<32><<<1, 32, 0, stream>>>(psum, psq, bn1_g, bn1_b, bsc, bsh);
  bn_relu_kernel<32><<<IMG * 32, 256, 0, stream>>>(r1, bsc, bsh);

  // ---- conv2 + BN2 + ReLU ----
  conv_wmma_kernel<32, 64, 9, false><<<IMG, 256, 0, stream>>>(r1, ap2, r2);
  reduce_stats_kernel<64><<<64 * 64, 256, 0, stream>>>(r2, psum, psq);
  finalize_stats_kernel<64><<<1, 64, 0, stream>>>(psum, psq, bn2_g, bn2_b, bsc, bsh);
  bn_relu_kernel<64><<<IMG * 64, 256, 0, stream>>>(r2, bsc, bsh);

  // ---- conv3 + BN3 + ReLU ----
  conv_wmma_kernel<64, 128, 18, false><<<IMG, 256, 0, stream>>>(r2, ap3, r3);
  reduce_stats_kernel<128><<<128 * 64, 256, 0, stream>>>(r3, psum, psq);
  finalize_stats_kernel<128><<<1, 128, 0, stream>>>(psum, psq, bn3_g, bn3_b, bsc, bsh);
  bn_relu_kernel<128><<<IMG * 128, 256, 0, stream>>>(r3, bsc, bsh);

  // ---- head ----
  pool_kernel<<<IMG, 128, 0, stream>>>(r3, feat);
  gemm_wmma_kernel<4, 16, 256, true><<<(256 * 16) / 8, 256, 0, stream>>>(feat, bfc, fc_b, hfeat);
  gemm_wmma_kernel<8, 32, 512, false><<<(256 * 32) / 8, 256, 0, stream>>>(hfeat, bg1, g1_b, g1out);
  ln_relu_kernel<<<IMG, 256, 0, stream>>>(g1out, ln1_g, ln1_b, h1);
  gemm_wmma_kernel<16, 32, 512, false><<<(256 * 32) / 8, 256, 0, stream>>>(h1, bg2, g2_b, g2out);
  ln_relu_kernel<<<IMG, 256, 0, stream>>>(g2out, ln2_g, ln2_b, h2);
  gemm_wmma_kernel<16, 1, 8, false><<<(256 * 1) / 8, 256, 0, stream>>>(h2, bg3, g3_b, g3out);

  // ---- softmax / top-k / gather ----
  softmax_kernel<<<16, 256, 0, stream>>>(g3out, out_assign, wt);
  topk_kernel<<<128, 256, 0, stream>>>(wt, topv, topi);
  gather_kernel<<<4096, 256, 0, stream>>>(x, topv, topi, out_main);
}
